// DiffeqSolverDecoder_31756988186965
// MI455X (gfx1250) — matrix operand order repaired and checked by reading, not compiled, not run on hardware
//
#include <hip/hip_runtime.h>
#include <math.h>

typedef _Float16 h16 __attribute__((ext_vector_type(16)));
typedef float    f8  __attribute__((ext_vector_type(8)));

#define WMMA_F16(a, b, c) \
  __builtin_amdgcn_wmma_f32_16x16x32_f16(false, (a), false, (b), (short)0, (c), false, false)

#define S_   3
#define B_   1024
#define D_   64
#define H_   128
#define O_   128
#define T_   64
#define ROWS (S_ * B_)            // 3072
#define NTOT (ROWS * D_)          // 196608
#define NBLK (ROWS / 16)          // 192 persistent workgroups
#define MAX_INNER 12
#define RTOL_ 1e-4f
#define ATOL_ 1e-5f

struct SolverState {
  float t, dt;
  unsigned accept;
  float sumsq;
  unsigned cnt, gen;
};

// Dormand-Prince 5(4) tableau
__constant__ float dcA[7][6] = {
  {0.f, 0.f, 0.f, 0.f, 0.f, 0.f},
  {0.2f, 0.f, 0.f, 0.f, 0.f, 0.f},
  {3.f/40.f, 9.f/40.f, 0.f, 0.f, 0.f, 0.f},
  {44.f/45.f, -56.f/15.f, 32.f/9.f, 0.f, 0.f, 0.f},
  {19372.f/6561.f, -25360.f/2187.f, 64448.f/6561.f, -212.f/729.f, 0.f, 0.f},
  {9017.f/3168.f, -355.f/33.f, 46732.f/5247.f, 49.f/176.f, -5103.f/18656.f, 0.f},
  {0.f, 0.f, 0.f, 0.f, 0.f, 0.f}
};
__constant__ float dcB[6] = {35.f/384.f, 0.f, 500.f/1113.f, 125.f/192.f, -2187.f/6784.f, 11.f/84.f};
__constant__ float dcE[7] = {71.f/57600.f, 0.f, -71.f/16695.f, 71.f/1920.f,
                             -17253.f/339200.f, 22.f/525.f, -1.f/40.f};

// Sense-reversing grid barrier over the 192 co-resident persistent workgroups.
__device__ __forceinline__ void grid_barrier(SolverState* st) {
  __syncthreads();
  if (threadIdx.x == 0) {
    __threadfence();
    volatile unsigned* gen = &st->gen;
    unsigned g = *gen;
    if (atomicAdd(&st->cnt, 1u) == (unsigned)(NBLK - 1)) {
      st->cnt = 0u;
      __threadfence();
      *gen = g + 1u;
    } else {
      while (*gen == g) { __builtin_amdgcn_s_sleep(1); }
    }
    __threadfence();
  }
  __syncthreads();
}

__global__ void init_kernel(const float* __restrict__ y0,
                            const float* __restrict__ times,
                            float* __restrict__ sol, SolverState* st) {
  long idx = (long)blockIdx.x * blockDim.x + threadIdx.x;
  if (idx < NTOT) {
    long sb = idx >> 6;
    int d = (int)(idx & 63);
    sol[sb * (long)(T_ * D_) + d] = y0[idx];   // sol_z[:, :, 0, :]
  }
  if (idx == 0) {
    st->t = times[0];
    st->dt = times[1] - times[0];
    st->accept = 0u;
    st->sumsq = 0.f;
    st->cnt = 0u;
    st->gen = 0u;
  }
}

// Persistent solver: one workgroup owns 16 rows; state lives in LDS for the whole solve.
__global__ __launch_bounds__(128)
void solve_kernel(const float* __restrict__ y0, const float* __restrict__ times,
                  const float* __restrict__ W1, const float* __restrict__ b1,
                  const float* __restrict__ W2, const float* __restrict__ b2,
                  float* __restrict__ sol, SolverState* st) {
  __shared__ float     sY[16][64];
  __shared__ float     sY5[16][64];
  __shared__ float     sK[7][16][64];
  __shared__ _Float16  sZ[16][72];     // f16 stage input, padded
  __shared__ _Float16  sH[16][136];    // f16 hidden, padded
  __shared__ float     sRed;

  const int tid = threadIdx.x;
  const int w = tid >> 5;
  const int lane = tid & 31;
  const int m = lane & 15;
  const int g = lane >> 4;
  const int row0 = blockIdx.x * 16;

  // --- Pin weight B-fragments in VGPRs (contiguous k order, same as A: k = 32*kc + 16*g + i, n = lane%16 + 16*nt)
  h16 B1f[2][2], B2f[4];
  #pragma unroll
  for (int kc = 0; kc < 2; ++kc) {
    #pragma unroll
    for (int i = 0; i < 16; ++i) {
      int k = kc * 32 + g * 16 + i;
      B1f[0][kc][i] = (_Float16)W1[k * H_ + w * 16 + m];
      B1f[1][kc][i] = (_Float16)W1[k * H_ + (w + 4) * 16 + m];
    }
  }
  #pragma unroll
  for (int kc = 0; kc < 4; ++kc) {
    #pragma unroll
    for (int i = 0; i < 16; ++i) {
      int k = kc * 32 + g * 16 + i;
      B2f[kc][i] = (_Float16)W2[k * D_ + w * 16 + m];
    }
  }
  const float b1v0 = b1[w * 16 + m];
  const float b1v1 = b1[(w + 4) * 16 + m];
  const float b2v  = b2[w * 16 + m];

  // Load initial state tile
  for (int e = tid; e < 1024; e += 128) {
    int r = e >> 6, c = e & 63;
    sY[r][c] = y0[(row0 + r) * D_ + c];
  }
  __syncthreads();

  // f(z) = tanh(z@W1+b1)@W2+b2 on the 16-row tile; result into sK[sidx]
  auto eval_f = [&](int sidx) {
    f8 a0 = {}; f8 a1 = {};
    #pragma unroll
    for (int kc = 0; kc < 2; ++kc) {
      h16 af;
      int base = kc * 32 + g * 16;
      #pragma unroll
      for (int i = 0; i < 8; ++i) { af[i] = sZ[m][base + i]; af[i + 8] = sZ[m][base + 8 + i]; }
      a0 = WMMA_F16(af, B1f[0][kc], a0);
      a1 = WMMA_F16(af, B1f[1][kc], a1);
    }
    #pragma unroll
    for (int r = 0; r < 8; ++r) {
      sH[r + 8 * g][w * 16 + m]       = (_Float16)tanhf(a0[r] + b1v0);
      sH[r + 8 * g][(w + 4) * 16 + m] = (_Float16)tanhf(a1[r] + b1v1);
    }
    __syncthreads();
    f8 acc = {};
    #pragma unroll
    for (int kc = 0; kc < 4; ++kc) {
      h16 af;
      int base = kc * 32 + g * 16;
      #pragma unroll
      for (int i = 0; i < 8; ++i) { af[i] = sH[m][base + i]; af[i + 8] = sH[m][base + 8 + i]; }
      acc = WMMA_F16(af, B2f[kc], acc);
    }
    #pragma unroll
    for (int r = 0; r < 8; ++r)
      sK[sidx][r + 8 * g][w * 16 + m] = acc[r] + b2v;
    __syncthreads();
  };

  volatile SolverState* vst = st;

  for (int iv = 0; iv < T_ - 1; ++iv) {
    const float t_target = times[iv + 1];
    for (int j = 0; j < MAX_INNER; ++j) {
      const float t  = vst->t;
      const float dt = vst->dt;
      const float remaining = t_target - t;
      const float dtq = fminf(dt, remaining);   // dt_try (may be <=0 when done, as in ref)

      // stage 0: z = y
      for (int e = tid; e < 1024; e += 128) {
        int r = e >> 6, c = e & 63;
        sZ[r][c] = (_Float16)sY[r][c];
      }
      __syncthreads();
      eval_f(0);

      // stages 1..5
      for (int s = 1; s <= 5; ++s) {
        for (int e = tid; e < 1024; e += 128) {
          int r = e >> 6, c = e & 63;
          float acc = 0.f;
          for (int q = 0; q < s; ++q) acc += dcA[s][q] * sK[q][r][c];
          sZ[r][c] = (_Float16)(sY[r][c] + dtq * acc);
        }
        __syncthreads();
        eval_f(s);
      }

      // 5th-order solution y5; k7 = f(y5)
      for (int e = tid; e < 1024; e += 128) {
        int r = e >> 6, c = e & 63;
        float acc = 0.f;
        #pragma unroll
        for (int q = 0; q < 6; ++q) acc += dcB[q] * sK[q][r][c];
        float y5v = sY[r][c] + dtq * acc;
        sY5[r][c] = y5v;
        sZ[r][c] = (_Float16)y5v;
      }
      __syncthreads();
      eval_f(6);

      // embedded error -> partial sum of (err/scale)^2
      if (tid == 0) sRed = 0.f;
      __syncthreads();
      float local = 0.f;
      for (int e = tid; e < 1024; e += 128) {
        int r = e >> 6, c = e & 63;
        float errv = 0.f;
        #pragma unroll
        for (int q = 0; q < 7; ++q) errv += dcE[q] * sK[q][r][c];
        errv *= dtq;
        float yv = sY[r][c], y5v = sY5[r][c];
        float scale = ATOL_ + RTOL_ * fmaxf(fabsf(yv), fabsf(y5v));
        float rr = errv / scale;
        local += rr * rr;
      }
      atomicAdd(&sRed, local);
      __syncthreads();
      if (tid == 0) atomicAdd(&st->sumsq, sRed);

      grid_barrier(st);

      if (blockIdx.x == 0 && tid == 0) {
        float ss = vst->sumsq;
        float ratio = sqrtf(ss / (float)NTOT);
        bool done = remaining <= 0.f;
        bool acc = (ratio <= 1.f) && !done;
        // snap t to t_target on last inner iteration (solve_to return), else scan update
        vst->t = (j == MAX_INNER - 1) ? t_target : (acc ? t + dtq : t);
        float factor = fminf(fmaxf(0.9f * powf(fmaxf(ratio, 1e-10f), -0.2f), 0.2f), 10.f);
        vst->dt = done ? dt : dt * factor;
        vst->accept = acc ? 1u : 0u;
        vst->sumsq = 0.f;
        __threadfence();
      }

      grid_barrier(st);

      if (vst->accept != 0u) {
        for (int e = tid; e < 1024; e += 128) {
          int r = e >> 6, c = e & 63;
          sY[r][c] = sY5[r][c];
        }
      }
      __syncthreads();
    }

    // snapshot committed state: sol_z[:, :, iv+1, :]
    for (int e = tid; e < 1024; e += 128) {
      int r = e >> 6, c = e & 63;
      sol[((long)(row0 + r) * T_ + (iv + 1)) * D_ + c] = sY[r][c];
    }
  }
}

// pred_x = sol_z @ Wo + bo  : [ROWS*T, 64] x [64, 128]
__global__ __launch_bounds__(128)
void decode_kernel(const float* __restrict__ sol, const float* __restrict__ Wo,
                   const float* __restrict__ bo, float* __restrict__ pred) {
  const int tid = threadIdx.x;
  const int w = tid >> 5;
  const int lane = tid & 31;
  const int m = lane & 15;
  const int g = lane >> 4;
  const long row0 = (long)blockIdx.x * 16;

  h16 Bf[2][2];
  #pragma unroll
  for (int kc = 0; kc < 2; ++kc) {
    #pragma unroll
    for (int i = 0; i < 16; ++i) {
      int k = kc * 32 + g * 16 + i;
      Bf[0][kc][i] = (_Float16)Wo[k * O_ + w * 16 + m];
      Bf[1][kc][i] = (_Float16)Wo[k * O_ + (w + 4) * 16 + m];
    }
  }
  const float bo0 = bo[w * 16 + m];
  const float bo1 = bo[(w + 4) * 16 + m];

  f8 a0 = {}, a1 = {};
  const float* rowp = sol + (row0 + m) * (long)D_;
  #pragma unroll
  for (int kc = 0; kc < 2; ++kc) {
    h16 af;
    int base = kc * 32 + g * 16;
    #pragma unroll
    for (int i = 0; i < 8; ++i) {
      af[i]     = (_Float16)rowp[base + i];
      af[i + 8] = (_Float16)rowp[base + 8 + i];
    }
    a0 = WMMA_F16(af, Bf[0][kc], a0);
    a1 = WMMA_F16(af, Bf[1][kc], a1);
  }
  #pragma unroll
  for (int r = 0; r < 8; ++r) {
    long rr = row0 + r + 8 * g;
    pred[rr * O_ + w * 16 + m]       = a0[r] + bo0;
    pred[rr * O_ + (w + 4) * 16 + m] = a1[r] + bo1;
  }
}

extern "C" void kernel_launch(void* const* d_in, const int* in_sizes, int n_in,
                              void* d_out, int out_size, void* d_ws, size_t ws_size,
                              hipStream_t stream) {
  const float* y0    = (const float*)d_in[0];
  const float* times = (const float*)d_in[1];
  const float* W1    = (const float*)d_in[2];
  const float* b1    = (const float*)d_in[3];
  const float* W2    = (const float*)d_in[4];
  const float* b2    = (const float*)d_in[5];
  const float* Wo    = (const float*)d_in[6];
  const float* bo    = (const float*)d_in[7];

  float* sol  = (float*)d_out;                       // sol_z: [S,B,T,D]
  float* pred = sol + (size_t)ROWS * T_ * D_;        // pred_x: [S,B,T,O]
  SolverState* st = (SolverState*)d_ws;

  init_kernel<<<(NTOT + 255) / 256, 256, 0, stream>>>(y0, times, sol, st);
  solve_kernel<<<NBLK, 128, 0, stream>>>(y0, times, W1, b1, W2, b2, sol, st);
  decode_kernel<<<(ROWS * T_) / 16, 128, 0, stream>>>(sol, Wo, bo, pred);
}
